// Mol_tox_GAT_15771119911194
// MI455X (gfx1250) — matrix operand order repaired and checked by reading, not compiled, not run on hardware
//
#include <hip/hip_runtime.h>
#include <hip/hip_bf16.h>
#include <stdint.h>

// ---------------- problem constants (match reference) ----------------
#define NNODE  50000          // 3125 * 16 -> no M padding needed
#define NEDGE  400000
#define NGRAPH 2048           // 128 * 16
#define NHEAD  4
#define NCH    64
#define HC     256            // NHEAD*NCH
#define NIN    29
#define NED    6
#define NTASK  12

typedef __attribute__((ext_vector_type(16))) _Float16 v16h;
typedef __attribute__((ext_vector_type(8)))  _Float16 h8;
typedef __attribute__((ext_vector_type(8)))  float    v8f;
typedef __attribute__((ext_vector_type(4)))  unsigned v4u;
typedef __attribute__((ext_vector_type(8)))  int      v8i;
typedef __attribute__((ext_vector_type(4)))  int      v4i;

// ---------------- helpers ----------------
static __device__ __forceinline__ unsigned f2ord(float f) {
  unsigned u = __float_as_uint(f);
  return (u & 0x80000000u) ? ~u : (u | 0x80000000u);
}
static __device__ __forceinline__ float ord2f(unsigned k) {
  unsigned u = (k & 0x80000000u) ? (k & 0x7fffffffu) : ~k;
  return __uint_as_float(u);
}

// =====================================================================
// Tensor Data Mover: 2-D tile (rows x rowLen halfs, contiguous) -> LDS
// D# built per CDNA5 ISA §8.3/8.4. Falls back to plain copies if absent.
// =====================================================================
#if __has_builtin(__builtin_amdgcn_tensor_load_to_lds)
#define HAVE_TDM 1
static __device__ __forceinline__ void tdm_load_2d(const void* gaddr, void* laddr,
                                                   int rowLenHalves, int rows) {
  unsigned long long ga = (unsigned long long)(uintptr_t)gaddr;
  unsigned lds_off = (unsigned)(uintptr_t)laddr;    // low 32 bits of flat LDS addr = offset
  v4u g0;
  g0.x = 0x1u;                                       // count=1, user mode, no gather
  g0.y = lds_off;                                    // lds_addr [63:32]
  g0.z = (unsigned)ga;                               // global_addr low   [95:64]
  g0.w = ((unsigned)(ga >> 32) & 0x01FFFFFFu)        // global_addr [120:96]
       | (2u << 30);                                 // type=2 ("image")  [127:126]
  unsigned d0 = (unsigned)rowLenHalves;              // tensor_dim0 (elements)
  unsigned d1 = (unsigned)rows;                      // tensor_dim1
  v8i g1;
  g1[0] = (int)(1u << 16);                           // wg_mask=0, data_size=1 (2 bytes)
  g1[1] = (int)((d0 & 0xFFFFu) << 16);               // [47:32]=0, tensor_dim0 lo16
  g1[2] = (int)(((d0 >> 16) & 0xFFFFu) | ((d1 & 0xFFFFu) << 16));   // dim0 hi | dim1 lo
  g1[3] = (int)(((d1 >> 16) & 0xFFFFu) | ((d0 & 0xFFFFu) << 16));   // dim1 hi | tile_dim0
  g1[4] = (int)(d1 & 0xFFFFu);                       // tile_dim1=rows, tile_dim2=0
  g1[5] = (int)d0;                                   // tensor_dim0_stride lo32 (= rowLen)
  g1[6] = 0;                                         // stride hi | dim1_stride lo
  g1[7] = 0;
  v4i g2 = {0, 0, 0, 0};
  v4i g3 = {0, 0, 0, 0};
#if __has_include(<hip/amd_detail/amd_gfx1250_TDM.h>)
  v8i g4 = {0, 0, 0, 0, 0, 0, 0, 0};                 // clang-23 / therock: 6-arg form
  __builtin_amdgcn_tensor_load_to_lds(g0, g1, g2, g3, g4, 0);
#else
  __builtin_amdgcn_tensor_load_to_lds(g0, g1, g2, g3, 0);   // ROCm 7.2: 5-arg form
#endif
}
#endif

// ---------------- generic fill ----------------
__global__ void fill_u32(unsigned* __restrict__ p, unsigned v, size_t n) {
  size_t i  = (size_t)blockIdx.x * blockDim.x + threadIdx.x;
  size_t st = (size_t)gridDim.x * blockDim.x;
  for (; i < n; i += st) p[i] = v;
}

// ---------------- mean of edge_attr columns (deterministic) ----------------
__global__ void edge_attr_mean(const float* __restrict__ ea, float* __restrict__ meanE) {
  __shared__ float red[256];
  const int j = blockIdx.x;              // 0..NED-1
  float s = 0.f;
  for (int e = threadIdx.x; e < NEDGE; e += blockDim.x)
    s += ea[(size_t)e * NED + j];
  red[threadIdx.x] = s;
  __syncthreads();
  for (int st = 128; st > 0; st >>= 1) {
    if ((int)threadIdx.x < st) red[threadIdx.x] += red[threadIdx.x + st];
    __syncthreads();
  }
  if (threadIdx.x == 0) meanE[j] = red[0] / (float)NEDGE;
}

// ---------------- pad/convert: f32 [M,K] -> f16 [M,Kp] zero-filled ----------------
__global__ void pad_convert_a(const float* __restrict__ A, _Float16* __restrict__ Ap,
                              int M, int K, int Kp) {
  size_t t = (size_t)blockIdx.x * blockDim.x + threadIdx.x;
  if (t >= (size_t)M * Kp) return;
  int m = (int)(t / Kp), k = (int)(t % Kp);
  float v = (k < K) ? A[(size_t)m * K + k] : 0.f;
  Ap[t] = (_Float16)v;
}

// ---------------- pad/convert/transpose: f32 [K,Nc] -> f16 [Np,Kp] ----------------
__global__ void pad_convert_bt(const float* __restrict__ B, _Float16* __restrict__ Bt,
                               int K, int Nc, int Np, int Kp) {
  int t = blockIdx.x * blockDim.x + threadIdx.x;
  if (t >= Np * Kp) return;
  int n = t / Kp, k = t % Kp;
  float v = (n < Nc && k < K) ? B[(size_t)k * Nc + n] : 0.f;
  Bt[t] = (_Float16)v;
}

// =====================================================================
// WMMA GEMM: C[M,Nc] = Apad[M,KP](f16) @ Bt[Np,KP]^T(f16)
// Block: 8 waves; one 16-row m-tile, 8 consecutive 16-col n-tiles.
// A tile + B slab staged to LDS via TDM (fallback: vector copies),
// fragments via ds_load_b128; K fully unrolled -> back-to-back v_wmma.
// M must be a multiple of 16 (true for 50000 and 2048).
// =====================================================================
template <int KP, bool GUARD_N, bool BIAS>
__global__ void __launch_bounds__(256) wmma_gemm(const _Float16* __restrict__ Apad,
                                                 const _Float16* __restrict__ Bt,
                                                 const float* __restrict__ bias,
                                                 float* __restrict__ C,
                                                 int Nc, int Np) {
  constexpr int NT_BLK = 8;                       // n-tiles per block
  __shared__ alignas(16) _Float16 ldsA[16 * KP];
  __shared__ alignas(16) _Float16 ldsB[NT_BLK * 16 * KP];

  const int lane = threadIdx.x & 31;
  const int wib  = threadIdx.x >> 5;
  const int tmBase  = blockIdx.x * 16;
  const int colBase = blockIdx.y * (NT_BLK * 16);
  const int ntiles  = (Nc + 15) >> 4;
  const int rowsB   = min(NT_BLK * 16, Np - colBase);

  const _Float16* gA = Apad + (size_t)tmBase * KP;
  const _Float16* gB = Bt + (size_t)colBase * KP;

#ifdef HAVE_TDM
  if (wib == 0) {
    tdm_load_2d(gA, ldsA, KP, 16);
    tdm_load_2d(gB, ldsB, KP, rowsB);
    __builtin_amdgcn_s_wait_tensorcnt(0);
  }
#else
  {
    const int nA = 16 * KP / 8;                   // h8 chunks
    const int nB = rowsB * KP / 8;
    for (int i = threadIdx.x; i < nA; i += 256)
      ((h8*)ldsA)[i] = ((const h8*)gA)[i];
    for (int i = threadIdx.x; i < nB; i += 256)
      ((h8*)ldsB)[i] = ((const h8*)gB)[i];
  }
#endif
  __syncthreads();

  const int tn = (colBase >> 4) + wib;
  if (tn < ntiles) {                              // wave-uniform: EXEC stays all-ones
    const int half = lane >> 4;
    const int l16  = lane & 15;
    const _Float16* la = ldsA + (size_t)l16 * KP;
    const _Float16* lb = ldsB + (size_t)(wib * 16 + l16) * KP;
    v8f acc = {};
#pragma unroll
    for (int k0 = 0; k0 < KP; k0 += 32) {
      // A 16x32 f16 layout: e<8 -> k=k0+8*half+e ; e>=8 -> k=k0+16+8*half+(e-8)
      h8 a0 = *(const h8*)(la + k0 + half * 8);
      h8 a1 = *(const h8*)(la + k0 + 16 + half * 8);
      // B 32x16 f16 layout (transposed weights): e<8 -> k=k0+8*half+e ; e>=8 -> k=k0+16+8*half+(e-8)
      h8 b0 = *(const h8*)(lb + k0 + half * 8);
      h8 b1 = *(const h8*)(lb + k0 + 16 + half * 8);
      v16h af, bf;
#pragma unroll
      for (int e = 0; e < 8; ++e) {
        af[e] = a0[e]; af[e + 8] = a1[e];
        bf[e] = b0[e]; bf[e + 8] = b1[e];
      }
      acc = __builtin_amdgcn_wmma_f32_16x16x32_f16(false, af, false, bf,
                                                   (short)0, acc, false, false);
    }
    // branch-free epilogue: bias hoisted, single per-lane column predicate
    const int col = tn * 16 + l16;
    const bool cok = !GUARD_N || (col < Nc);
    float bv = 0.f;
    if (BIAS) bv = cok ? bias[col] : 0.f;
#pragma unroll
    for (int r = 0; r < 8; ++r) {                 // C/D: VGPR r -> row r + 8*half
      const int m = tmBase + r + half * 8;
      const float o = BIAS ? (acc[r] + bv) : acc[r];
      if (cok) C[(size_t)m * Nc + col] = o;
    }
  }
}

// ---------------- per-node attention logits al_s, al_d ----------------
__global__ void node_attn_logits(const float* __restrict__ h,
                                 const float* __restrict__ aS, const float* __restrict__ aD,
                                 float* __restrict__ alS, float* __restrict__ alD) {
  int t = blockIdx.x * blockDim.x + threadIdx.x;
  if (t >= NNODE * NHEAD) return;
  int n = t >> 2, hd = t & 3;
  const float* hp = h + (size_t)n * HC + hd * NCH;
  float s = 0.f, d = 0.f;
#pragma unroll 8
  for (int c = 0; c < NCH; ++c) {
    float v = hp[c];
    s += v * aS[hd * NCH + c];
    d += v * aD[hd * NCH + c];
  }
  alS[t] = s;
  alD[t] = d;
}

// ---------------- tiny edge projection: V[6][4], loopAle[4] ----------------
__global__ void prep_edge_proj(const float* __restrict__ We, const float* __restrict__ ae,
                               const float* __restrict__ meanE,
                               float* __restrict__ V, float* __restrict__ loopAle) {
  int t = threadIdx.x;
  if (t < NED * NHEAD) {
    int j = t >> 2, hd = t & 3;
    float s = 0.f;
    for (int c = 0; c < NCH; ++c)
      s += We[(size_t)j * HC + hd * NCH + c] * ae[hd * NCH + c];
    V[j * NHEAD + hd] = s;
  }
  __syncthreads();
  if (t < NHEAD) {
    float s = 0.f;
    for (int j = 0; j < NED; ++j) s += meanE[j] * V[j * NHEAD + t];
    loopAle[t] = s;
  }
}

// ---------------- pass 1: leaky-relu logits + segment max ----------------
__global__ void edge_logit_max(const int* __restrict__ ei, const float* __restrict__ eattr,
                               const float* __restrict__ alS, const float* __restrict__ alD,
                               const float* __restrict__ V, const float* __restrict__ loopAle,
                               float* __restrict__ logitB, unsigned* __restrict__ amaxK) {
  const int ETOT = NEDGE + NNODE;
  int e = blockIdx.x * blockDim.x + threadIdx.x;
  if (e >= ETOT) return;
  int s, d;
  float ale[NHEAD];
  if (e < NEDGE) {
    s = ei[e];
    d = ei[NEDGE + e];
    float ea[NED];
#pragma unroll
    for (int j = 0; j < NED; ++j) ea[j] = eattr[(size_t)e * NED + j];
#pragma unroll
    for (int hd = 0; hd < NHEAD; ++hd) {
      float t = 0.f;
#pragma unroll
      for (int j = 0; j < NED; ++j) t += ea[j] * V[j * NHEAD + hd];
      ale[hd] = t;
    }
  } else {
    s = d = e - NEDGE;
#pragma unroll
    for (int hd = 0; hd < NHEAD; ++hd) ale[hd] = loopAle[hd];
  }
#pragma unroll
  for (int hd = 0; hd < NHEAD; ++hd) {
    float a = alS[s * NHEAD + hd] + alD[d * NHEAD + hd] + ale[hd];
    a = (a > 0.f) ? a : 0.2f * a;                 // leaky_relu(0.2)
    logitB[(size_t)e * NHEAD + hd] = a;
    atomicMax(&amaxK[d * NHEAD + hd], f2ord(a));
  }
}

// ---------------- pass 2: ex = exp(logit - amax); denom += ex ----------------
__global__ void edge_softmax_sum(const int* __restrict__ ei, const unsigned* __restrict__ amaxK,
                                 float* __restrict__ logitB, float* __restrict__ denom) {
  const int ETOT = NEDGE + NNODE;
  int e = blockIdx.x * blockDim.x + threadIdx.x;
  if (e >= ETOT) return;
  int d = (e < NEDGE) ? ei[NEDGE + e] : (e - NEDGE);
#pragma unroll
  for (int hd = 0; hd < NHEAD; ++hd) {
    float am = ord2f(amaxK[d * NHEAD + hd]);
    float ex = __expf(logitB[(size_t)e * NHEAD + hd] - am);
    logitB[(size_t)e * NHEAD + hd] = ex;
    atomicAdd(&denom[d * NHEAD + hd], ex);
  }
}

// ---------------- pass 3: acc[dst] += ex * h[src] ----------------
__global__ void edge_message(const int* __restrict__ ei, const float* __restrict__ h,
                             const float* __restrict__ exB, float* __restrict__ acc) {
  const size_t ETOT = NEDGE + NNODE;
  size_t t = (size_t)blockIdx.x * blockDim.x + threadIdx.x;
  if (t >= ETOT * NCH) return;
  size_t e = t >> 6;
  int c = (int)(t & 63);
  int s, d;
  if (e < NEDGE) { s = ei[e]; d = ei[NEDGE + e]; }
  else           { s = d = (int)(e - NEDGE); }
  const float* hp = h + (size_t)s * HC;
  float* ap = acc + (size_t)d * HC;
#pragma unroll
  for (int hd = 0; hd < NHEAD; ++hd) {
    float ex = exB[e * NHEAD + hd];
    atomicAdd(&ap[hd * NCH + c], hp[hd * NCH + c] * ex);
  }
}

// ---------------- pass 4: out = relu(acc/denom + bias); also emit f16 copy ----------------
__global__ void node_finalize(const float* __restrict__ acc, const float* __restrict__ denom,
                              const float* __restrict__ bias, float* __restrict__ out,
                              _Float16* __restrict__ outH) {
  size_t t = (size_t)blockIdx.x * blockDim.x + threadIdx.x;
  if (t >= (size_t)NNODE * HC) return;
  int n  = (int)(t >> 8);
  int hc = (int)(t & 255);
  int hd = hc >> 6;
  float v = acc[t] / (denom[n * NHEAD + hd] + 1e-16f) + bias[hc];
  v = fmaxf(v, 0.f);
  out[t]  = v;
  outH[t] = (_Float16)v;      // f16 operand for next layer's WMMA GEMM
}

// ---------------- pooling ----------------
__global__ void pool_accumulate(const float* __restrict__ feat, const int* __restrict__ batch,
                                float* __restrict__ sums, float* __restrict__ cnt) {
  size_t t = (size_t)blockIdx.x * blockDim.x + threadIdx.x;
  if (t >= (size_t)NNODE * HC) return;
  int n  = (int)(t >> 8);
  int hc = (int)(t & 255);
  int g  = batch[n];
  atomicAdd(&sums[(size_t)g * HC + hc], feat[t]);
  if (hc == 0) atomicAdd(&cnt[g], 1.0f);
}

__global__ void pool_divide(float* __restrict__ sums, const float* __restrict__ cnt,
                            _Float16* __restrict__ outH) {
  size_t t = (size_t)blockIdx.x * blockDim.x + threadIdx.x;
  if (t >= (size_t)NGRAPH * HC) return;
  int g = (int)(t >> 8);
  float v = sums[t] / fmaxf(cnt[g], 1.0f);
  sums[t] = v;
  outH[t] = (_Float16)v;      // f16 operand for readout WMMA GEMM
}

// ---------------- host side ----------------
extern "C" void kernel_launch(void* const* d_in, const int* in_sizes, int n_in,
                              void* d_out, int out_size, void* d_ws, size_t ws_size,
                              hipStream_t stream) {
  (void)in_sizes; (void)n_in; (void)out_size; (void)ws_size;

  const float* x     = (const float*)d_in[0];
  const int*   ei    = (const int*)  d_in[1];
  const float* eattr = (const float*)d_in[2];
  const int*   batch = (const int*)  d_in[3];
  const float* W1  = (const float*)d_in[4];
  const float* as1 = (const float*)d_in[5];
  const float* ad1 = (const float*)d_in[6];
  const float* We1 = (const float*)d_in[7];
  const float* ae1 = (const float*)d_in[8];
  const float* b1  = (const float*)d_in[9];
  const float* W2  = (const float*)d_in[10];
  const float* as2 = (const float*)d_in[11];
  const float* ad2 = (const float*)d_in[12];
  const float* We2 = (const float*)d_in[13];
  const float* ae2 = (const float*)d_in[14];
  const float* b2  = (const float*)d_in[15];
  const float* Wl  = (const float*)d_in[16];
  const float* bl  = (const float*)d_in[17];
  float* out = (float*)d_out;

  // workspace carve-out (256B aligned slices)
  char* wp = (char*)d_ws;
  auto alloc = [&](size_t bytes) -> void* {
    void* p = (void*)wp;
    wp += (bytes + 255) & ~(size_t)255;
    return p;
  };
  const size_t ETOT = (size_t)NEDGE + NNODE;
  float*    bufA    = (float*)alloc(sizeof(float) * NNODE * HC);       // projection h
  float*    bufB    = (float*)alloc(sizeof(float) * NNODE * HC);       // aggregation acc
  float*    bufC    = (float*)alloc(sizeof(float) * NNODE * HC);       // activated features
  _Float16* Apad    = (_Float16*)alloc(sizeof(_Float16) * NNODE * HC); // f16 GEMM A operand
  _Float16* Btw     = (_Float16*)alloc(sizeof(_Float16) * HC * HC);    // f16 transposed W
  float*    alS     = (float*)alloc(sizeof(float) * NNODE * NHEAD);
  float*    alD     = (float*)alloc(sizeof(float) * NNODE * NHEAD);
  float*    logitB  = (float*)alloc(sizeof(float) * ETOT * NHEAD);
  unsigned* amaxK   = (unsigned*)alloc(sizeof(unsigned) * NNODE * NHEAD);
  float*    denom   = (float*)alloc(sizeof(float) * NNODE * NHEAD);
  float*    Vm      = (float*)alloc(sizeof(float) * NED * NHEAD);
  float*    loopAle = (float*)alloc(sizeof(float) * NHEAD);
  float*    meanE   = (float*)alloc(sizeof(float) * NED);
  float*    sums    = (float*)alloc(sizeof(float) * NGRAPH * HC);
  float*    cnt     = (float*)alloc(sizeof(float) * NGRAPH);

  // edge-attr mean (needed for self-loop edge features)
  edge_attr_mean<<<NED, 256, 0, stream>>>(eattr, meanE);

  auto run_layer = [&](int K, const float* W,
                       const float* aS, const float* aD,
                       const float* We, const float* ae, const float* b) {
    const int Kp = (K + 31) & ~31;                       // 32 or 256
    // stage transposed f16 weights (Apad already holds the f16 A operand)
    {
      int nb = HC * Kp;
      pad_convert_bt<<<(nb + 255) / 256, 256, 0, stream>>>(W, Btw, K, HC, HC, Kp);
    }
    // projection GEMM via WMMA (8 waves/block, 128 cols/block)
    dim3 grid(NNODE / 16, HC / 128);
    if (Kp == 32)
      wmma_gemm<32, false, false><<<grid, 256, 0, stream>>>(Apad, Btw, nullptr, bufA, HC, HC);
    else
      wmma_gemm<256, false, false><<<grid, 256, 0, stream>>>(Apad, Btw, nullptr, bufA, HC, HC);

    node_attn_logits<<<(NNODE * NHEAD + 255) / 256, 256, 0, stream>>>(bufA, aS, aD, alS, alD);
    prep_edge_proj<<<1, 32, 0, stream>>>(We, ae, meanE, Vm, loopAle);
    fill_u32<<<1024, 256, 0, stream>>>(amaxK, 0x00800000u /* ord(-FLT_MAX) */, (size_t)NNODE * NHEAD);
    fill_u32<<<1024, 256, 0, stream>>>((unsigned*)denom, 0u, (size_t)NNODE * NHEAD);
    fill_u32<<<8192, 256, 0, stream>>>((unsigned*)bufB, 0u, (size_t)NNODE * HC);
    int eb = (int)((ETOT + 255) / 256);
    edge_logit_max<<<eb, 256, 0, stream>>>(ei, eattr, alS, alD, Vm, loopAle, logitB, amaxK);
    edge_softmax_sum<<<eb, 256, 0, stream>>>(ei, amaxK, logitB, denom);
    {
      size_t t = ETOT * NCH;
      edge_message<<<(int)((t + 255) / 256), 256, 0, stream>>>(ei, bufA, logitB, bufB);
    }
    // finalize writes f32 features (bufC) + f16 GEMM operand (Apad) for the next layer
    node_finalize<<<(int)(((size_t)NNODE * HC + 255) / 256), 256, 0, stream>>>(bufB, denom, b,
                                                                               bufC, Apad);
  };

  // layer 1: stage f16 input (x, K=29 -> Kp=32)
  {
    size_t na = (size_t)NNODE * 32;
    pad_convert_a<<<(int)((na + 255) / 256), 256, 0, stream>>>(x, Apad, NNODE, NIN, 32);
  }
  run_layer(NIN, W1, as1, ad1, We1, ae1, b1);    // consumes Apad, refreshes Apad via finalize
  run_layer(HC,  W2, as2, ad2, We2, ae2, b2);

  // global mean pool (pool_divide also emits f16 operand into Apad)
  fill_u32<<<2048, 256, 0, stream>>>((unsigned*)sums, 0u, (size_t)NGRAPH * HC);
  fill_u32<<<32, 256, 0, stream>>>((unsigned*)cnt, 0u, (size_t)NGRAPH);
  pool_accumulate<<<(int)(((size_t)NNODE * HC + 255) / 256), 256, 0, stream>>>(bufC, batch, sums, cnt);
  pool_divide<<<(int)(((size_t)NGRAPH * HC + 255) / 256), 256, 0, stream>>>(sums, cnt, Apad);

  // readout: [G,256] @ [256,12] + bl  (Np padded to 16; guarded stores, hoisted bias)
  {
    int nb = 16 * HC;
    pad_convert_bt<<<(nb + 255) / 256, 256, 0, stream>>>(Wl, Btw, HC, NTASK, 16, HC);
    dim3 grid(NGRAPH / 16, 1);
    wmma_gemm<256, true, true><<<grid, 256, 0, stream>>>(Apad, Btw, bl, out, NTASK, 16);
  }
}
